// BMNN_53206054863098
// MI455X (gfx1250) — matrix operand order repaired and checked by reading, not compiled, not run on hardware
//
#include <hip/hip_runtime.h>
#include <hip/hip_bf16.h>
#include <math.h>

// CDNA5 / gfx1250: wave32, WMMA 16x16x32 bf16 (probe-confirmed builtin).
typedef __attribute__((ext_vector_type(16))) __bf16 v16bf;
typedef __attribute__((ext_vector_type(8)))  float  v8f;

#define HDIM 1536
#define WDIM 1536
#define NG   191      // reference grid points per axis: 0,8,...,1520
#define RMAX 1520     // last reference coordinate
#define OC   31       // max candidates per axis
#define SPAN 1528     // candidate top-left positions per axis: 0..1527
#define NBLK 96       // 16-wide candidate-b blocks (96*16 = 1536, tail masked)

// Zero the whole output first: masked/out-of-range candidate slots must be 0,
// and the main kernel scatters only valid (ref,candidate) pairs.
__global__ void bmnn_zero_out(float* __restrict__ out, int n) {
  int i = blockIdx.x * blockDim.x + threadIdx.x;
  if (i < n) out[i] = 0.f;
}

// One wave per (a, b-block) tile, BOTH ry-groups in the same wave so the B
// fragments (candidate pixels) and Sc are loaded/converted exactly once:
//   M = 16 reference patches (4 rx-slots x 4 ry-slots), per ry-group g=0/1
//   N = 16 candidates: fixed row a, columns b0..b0+15
//   K = 64 patch pixels -> two v_wmma_f32_16x16x32_bf16 per group (4 total)
// The two ry-groups cover the 6 ry grid points a 16-wide b block can touch;
// groups are disjoint, so every valid (ref,cand) pair is written exactly once.
__global__ __launch_bounds__(256) void bmnn_patch_dist(const float* __restrict__ img,
                                                       float* __restrict__ out) {
  const int lane = threadIdx.x & 31;
  const int wave = threadIdx.x >> 5;
  const int tile = blockIdx.x * 8 + wave;     // SPAN*NBLK = 146688 = 18336*8 exact

  const int a  = tile / NBLK;                 // candidate row (x) position
  const int b0 = (tile % NBLK) * 16;          // candidate column block base

  // Valid rx for a:  a-10 <= rx <= a+20 ;  valid ry for b: b-10 <= ry <= b+20
  const int rxs = (((a  - 10) + 7) >> 3) << 3;   // ceil((a-10)/8)*8 (arith shift)
  const int qy  = (((b0 - 10) + 7) >> 3) << 3;

  const int hi = (lane >> 4) & 1;   // half-wave: selects K-subset per ISA layout
  const int lm = lane & 15;

  // ---------------- B fragments: candidate patch pixels (shared by g=0,1) --
  // B (32x16 bf16, contiguous K order used here): lane<16 col N=lm holds K 0..15 (frag0) ->
  // patch rows i = {2hi, 2hi+1} (frag0) and {4+2hi, 5+2hi} (frag1), j=0..7.
  const int nb = b0 + lm;
  const int bc = nb > SPAN - 1 ? SPAN - 1 : nb;           // clamp (masked later)

  v16bf Bfr[2];
  float scPart = 0.f;
#pragma unroll
  for (int f = 0; f < 2; ++f) {
#pragma unroll
    for (int h = 0; h < 2; ++h) {
      const int irow = 4 * f + 2 * hi + h;
      const float* q = img + (a + irow) * WDIM + bc;
#pragma unroll
      for (int j = 0; j < 8; ++j) {
        __bf16 bv = (__bf16)q[j];          // round to bf16 (matches WMMA input)
        float  fb = (float)bv;
        scPart = fmaf(fb, fb, scPart);     // Sc from the *rounded* pixels
        Bfr[f][h * 8 + j] = bv;
      }
    }
  }
  // Lanes l and l^16 hold complementary halves of each candidate's 64 pixels.
  const float Sc  = scPart + __shfl_xor(scPart, 16, 32);   // full ||cand||^2
  const bool  bok = (nb < SPAN);

  // rx side is independent of the ry-group: compute once.
  const int mi = lm >> 2, mj = lm & 3;
  const int rx  = rxs + 8 * mi;
  const int rxc = rx < 0 ? 0 : (rx > RMAX ? RMAX : rx);    // clamp (masked later)

#pragma unroll
  for (int g = 0; g < 2; ++g) {
    const int ryb = qy + 32 * g;

    // ---------------- A fragments: reference patch pixels -----------------
    // A (16x32 bf16): same K order as B ->
    // patch rows i = {2hi, 2hi+1} (frag0) and {4+2hi, 5+2hi} (frag1), j=0..7.
    const int ry  = ryb + 8 * mj;
    const int ryc = ry < 0 ? 0 : (ry > RMAX ? RMAX : ry);

    v16bf Afr[2];
    float sbPart = 0.f;
#pragma unroll
    for (int f = 0; f < 2; ++f) {
#pragma unroll
      for (int h = 0; h < 2; ++h) {
        const int irow = 4 * f + 2 * hi + h;
        const float* p = img + (rxc + irow) * WDIM + ryc;
#pragma unroll
        for (int j = 0; j < 8; ++j) {
          __bf16 bv = (__bf16)p[j];
          float  fb = (float)bv;
          sbPart = fmaf(fb, fb, sbPart);   // Sb from the *rounded* pixels
          Afr[f][h * 8 + j] = bv;
        }
      }
    }

    // ---------------- corr = A x B via WMMA (K = 64 = 2 x 32) -------------
    v8f c = {0.f, 0.f, 0.f, 0.f, 0.f, 0.f, 0.f, 0.f};
    c = __builtin_amdgcn_wmma_f32_16x16x32_bf16(false, Afr[0], false, Bfr[0],
                                                (short)0, c, false, false);
    c = __builtin_amdgcn_wmma_f32_16x16x32_bf16(false, Afr[1], false, Bfr[1],
                                                (short)0, c, false, false);

    const float SbSelf = sbPart + __shfl_xor(sbPart, 16, 32);  // Sb[m = lm]

    // -------------- epilogue: d = sqrt(Sb + Sc - 2 corr), masked scatter --
    // C/D layout: lane<16 -> col N=lm, rows M=0..7 in c[0..7]; lane>=16 -> 8..15.
#pragma unroll
    for (int rr = 0; rr < 8; ++rr) {
      const int m = hi * 8 + rr;
      const float Sb = __shfl(SbSelf, m, 32);  // lane m (and m+16) holds Sb[m]
      const int mrx = rxs + 8 * (m >> 2);
      const int mry = ryb + 8 * (m & 3);
      const bool ok = bok
          && (mrx >= 0) && (mrx <= RMAX)
          && (mry >= 0) && (mry <= RMAX)
          && (mrx <= a + 20)                   // mrx >= a-10 holds by construction
          && (mry >= nb - 10) && (mry <= nb + 20);
      if (ok) {
        // all quantities below are provably non-negative -> unsigned 32-bit
        // index math (single v_mad chain + scale_offset store)
        const unsigned sx = (unsigned)(mrx - 20 > 0 ? mrx - 20 : 0);
        const unsigned sy = (unsigned)(mry - 20 > 0 ? mry - 20 : 0);
        const unsigned u = (unsigned)a  - sx;  // 0..30 when ok
        const unsigned v = (unsigned)nb - sy;
        const float d2 = fmaf(-2.f, c[rr], Sb + Sc);
        const unsigned idx =
            ((unsigned)(mrx >> 3) * NG + (unsigned)(mry >> 3)) * (OC * OC)
            + u * OC + v;
        // bf16 inputs bound accuracy at ~0.2%; hardware v_sqrt_f32 suffices.
        out[idx] = __builtin_amdgcn_sqrtf(fmaxf(d2, 0.f));
      }
    }
  }
}

extern "C" void kernel_launch(void* const* d_in, const int* in_sizes, int n_in,
                              void* d_out, int out_size, void* d_ws, size_t ws_size,
                              hipStream_t stream) {
  (void)in_sizes; (void)n_in; (void)d_ws; (void)ws_size;
  const float* img = (const float*)d_in[0];
  float* out = (float*)d_out;

  // 1) zero the output (reference zeroes masked candidate slots)
  bmnn_zero_out<<<(out_size + 255) / 256, 256, 0, stream>>>(out, out_size);

  // 2) WMMA patch-distance: 1528 a-rows * 96 b-blocks, 2 ry-groups per wave
  const int blocks = (SPAN * NBLK) / 8;       // 18336, exact (8 waves / block)
  bmnn_patch_dist<<<blocks, 256, 0, stream>>>(img, out);
}
